// GTransformerLayer_82059645157591
// MI455X (gfx1250) — compile-verified, matrix operand order checked by reading, not run on hardware
//
#include <hip/hip_runtime.h>
#include <hip/hip_bf16.h>

#define NN 16384
#define EE 262144
#define RR 5
#define HH 4
// D_IN = D_OUT = 128, d_k = 32

typedef __attribute__((ext_vector_type(16))) __bf16 v16bf;
typedef __attribute__((ext_vector_type(8)))  float  v8f;

// ---------------- WMMA fragment helper (CDNA5 wave32 layout) ----------------
// Loads a 16x32 bf16 fragment from a row-major tile with stride ld.
// For the A matrix (16x32, M x K): lane L -> row M = L%16;
//   lanes 0-15 hold K in {0..7, 16..23}, lanes 16-31 hold K in {8..15, 24..31}.
// For the B matrix (32x16, K x N) stored TRANSPOSED as [N][K]: lane L -> col
//   N = L%16 with the identical K pattern, so the same loader applies.
// K-pairs are contiguous -> each fragment is two 16-byte LDS loads per lane.
__device__ __forceinline__ v16bf load_frag(const __bf16* base, int ld) {
    const int lane = threadIdx.x & 31;
    const int m  = lane & 15;
    const int kb = (lane < 16) ? 0 : 8;
    v16bf f;
#pragma unroll
    for (int p = 0; p < 8; ++p) {
        const int k = ((p < 4) ? 0 : 16) + kb + 2 * (p & 3);
        f[2 * p]     = base[m * ld + k];
        f[2 * p + 1] = base[m * ld + k + 1];
    }
    return f;
}

// ---------------- conversions ----------------
__global__ void cvt_f32_bf16(const float* __restrict__ s, __bf16* __restrict__ d, int n) {
    for (int i = blockIdx.x * blockDim.x + threadIdx.x; i < n; i += gridDim.x * blockDim.x)
        d[i] = (__bf16)s[i];
}

// [mats][128 k][128 n] f32 -> [mats][128 n][128 k] bf16 (transpose for B-side)
__global__ void cvt_w_t(const float* __restrict__ s, __bf16* __restrict__ d, int mats) {
    const int total = mats * 128 * 128;
    for (int i = blockIdx.x * blockDim.x + threadIdx.x; i < total; i += gridDim.x * blockDim.x) {
        const int m = i / (128 * 128), rem = i % (128 * 128);
        const int n = rem >> 7, k = rem & 127;
        d[i] = (__bf16)s[(size_t)m * 16384 + k * 128 + n];
    }
}

// Wt [512 k][128 n] f32 -> Wt_t [128 n][512 k] bf16
__global__ void cvt_wt_t(const float* __restrict__ s, __bf16* __restrict__ d) {
    const int total = 128 * 512;
    for (int i = blockIdx.x * blockDim.x + threadIdx.x; i < total; i += gridDim.x * blockDim.x) {
        const int n = i >> 9, k = i & 511;
        d[i] = (__bf16)s[k * 128 + n];
    }
}

// ---------------- per-relation K/Q/V projections via WMMA ----------------
// grid.x = 3*RR*(NN/128); block = 256 (8 waves). Each block: 128x128 output tile.
__global__ void proj_gemm(const __bf16* __restrict__ hb,
                          const __bf16* __restrict__ Wbt,  // [3*RR][128 n][128 k] bf16
                          const float* __restrict__ bk,
                          const float* __restrict__ bq,
                          const float* __restrict__ bv,
                          float* __restrict__ KQV)         // [3*RR][NN][128] f32
{
    const int tilesPerN = NN / 128;
    const int tile = blockIdx.x % tilesPerN;
    const int pr   = blockIdx.x / tilesPerN;   // 0..14 : proj*RR + rel
    const int p    = pr / RR;
    const int r    = pr % RR;
    const int rowBase = tile * 128;

    const __bf16* W    = Wbt + (size_t)pr * 128 * 128;
    const float*  bias = ((p == 0) ? bk : (p == 1) ? bq : bv) + r * 128;

    __shared__ __bf16 As[128 * 128];   // 32 KB  [m][k]
    __shared__ __bf16 Bs[128 * 128];   // 32 KB  [n][k] (transposed W)

    // stage tiles into LDS (16B vector copies)
    {
        const uint4* srca = (const uint4*)(hb + (size_t)rowBase * 128);
        uint4*       dsta = (uint4*)As;
        const uint4* srcb = (const uint4*)W;
        uint4*       dstb = (uint4*)Bs;
        for (int i = threadIdx.x; i < 2048; i += blockDim.x) { dsta[i] = srca[i]; dstb[i] = srcb[i]; }
    }
    __syncthreads();

    const int wave = threadIdx.x >> 5;
    const int m0   = wave * 16;
    const int lane = threadIdx.x & 31;
    const int nsub = lane & 15;
    const int mo   = (lane < 16) ? 0 : 8;

    // cache the 4 A fragments for this wave's row stripe
    v16bf afrag[4];
#pragma unroll
    for (int kt = 0; kt < 4; ++kt)
        afrag[kt] = load_frag(As + m0 * 128 + kt * 32, 128);

    float* outBase = KQV + ((size_t)pr * NN + rowBase + m0) * 128;

#pragma unroll
    for (int nt = 0; nt < 8; ++nt) {
        v8f acc = {0.f, 0.f, 0.f, 0.f, 0.f, 0.f, 0.f, 0.f};
#pragma unroll
        for (int kt = 0; kt < 4; ++kt) {
            v16bf b = load_frag(Bs + (nt * 16) * 128 + kt * 32, 128);
            acc = __builtin_amdgcn_wmma_f32_16x16x32_bf16(false, afrag[kt], false, b,
                                                          (short)0, acc, false, false);
        }
        const float bv_ = bias[nt * 16 + nsub];
#pragma unroll
        for (int j = 0; j < 8; ++j)
            outBase[(size_t)(mo + j) * 128 + nt * 16 + nsub] = acc[j] + bv_;
    }
}

// ---------------- edge scores + segment max ----------------
__device__ __forceinline__ unsigned f32_ordered(float f) {
    unsigned u = __float_as_uint(f);
    return (u & 0x80000000u) ? ~u : (u | 0x80000000u);
}
__device__ __forceinline__ float ordered_f32(unsigned o) {
    unsigned u = (o & 0x80000000u) ? (o & 0x7FFFFFFFu) : ~o;
    return __uint_as_float(u);
}

__global__ void edge_scores(const float* __restrict__ KQV,
                            const int* __restrict__ src, const int* __restrict__ dst,
                            const int* __restrict__ etype,
                            float* __restrict__ score, unsigned* __restrict__ segmax)
{
    const int idx = blockIdx.x * blockDim.x + threadIdx.x;
    if (idx >= EE * HH) return;
    const int e = idx >> 2, head = idx & 3;
    const int s = src[e], d = dst[e], et = etype[e];
    const float4* k4 = (const float4*)(KQV + ((size_t)(0 * RR + et) * NN + s) * 128 + head * 32);
    const float4* q4 = (const float4*)(KQV + ((size_t)(1 * RR + et) * NN + d) * 128 + head * 32);
    float acc = 0.f;
#pragma unroll
    for (int i = 0; i < 8; ++i) {
        const float4 a = k4[i], b = q4[i];
        acc += a.x * b.x + a.y * b.y + a.z * b.z + a.w * b.w;
    }
    const float sc = acc * 0.17677669529663687f;   // 1/sqrt(32)
    score[idx] = sc;
    atomicMax(&segmax[((size_t)d * RR + et) * HH + head], f32_ordered(sc));
}

// ---------------- exp(score - max) and denominators ----------------
__global__ void edge_exp(const int* __restrict__ dst, const int* __restrict__ etype,
                         float* __restrict__ score,
                         const unsigned* __restrict__ segmax, float* __restrict__ den)
{
    const int idx = blockIdx.x * blockDim.x + threadIdx.x;
    if (idx >= EE * HH) return;
    const int e = idx >> 2, head = idx & 3;
    const int d = dst[e], et = etype[e];
    const size_t seg = ((size_t)d * RR + et) * HH + head;
    const float m = ordered_f32(segmax[seg]);
    const float ex = __expf(score[idx] - m);
    score[idx] = ex;                    // reuse buffer for exp values
    atomicAdd(&den[seg], ex);
}

// ---------------- weighted message aggregation ----------------
// thread per (edge, dim); loops 4 heads. agg layout: [NN][HH*128]
__global__ void edge_agg(const float* __restrict__ KQV,
                         const int* __restrict__ src, const int* __restrict__ dst,
                         const int* __restrict__ etype,
                         const float* __restrict__ exbuf, const float* __restrict__ den,
                         float* __restrict__ agg)
{
    const int idx = blockIdx.x * blockDim.x + threadIdx.x;
    if (idx >= EE * 128) return;
    const int e = idx >> 7, dim = idx & 127;
    const int s = src[e], d = dst[e], et = etype[e];
    const float v = KQV[((size_t)(2 * RR + et) * NN + s) * 128 + dim];
    const float* ex4  = exbuf + (size_t)e * HH;
    const float* den4 = den + ((size_t)d * RR + et) * HH;
#pragma unroll
    for (int head = 0; head < HH; ++head) {
        const float a = ex4[head] / den4[head];
        atomicAdd(&agg[(size_t)d * 512 + head * 128 + dim], a * v);
    }
}

// ---------------- output projection: [NN,512] @ [512,128] + bt ----------------
__global__ void out_gemm(const float* __restrict__ agg,
                         const __bf16* __restrict__ Wtt,  // [128 n][512 k] bf16
                         const float* __restrict__ bt,
                         float* __restrict__ out)
{
    const int rowBase = blockIdx.x * 128;
    __shared__ __bf16 As[128 * 128];   // 32 KB  [m][k-chunk]
    __shared__ __bf16 Bs[128 * 128];   // 32 KB  [n][k-chunk]

    const int wave = threadIdx.x >> 5;
    const int m0   = wave * 16;
    const int lane = threadIdx.x & 31;
    const int nsub = lane & 15;
    const int mo   = (lane < 16) ? 0 : 8;

    v8f acc[8];
#pragma unroll
    for (int nt = 0; nt < 8; ++nt) acc[nt] = (v8f){0.f,0.f,0.f,0.f,0.f,0.f,0.f,0.f};

    for (int kc = 0; kc < 4; ++kc) {
        // convert the agg chunk to bf16 in LDS, packed 2 values per 32-bit store
        {
            unsigned* As32 = (unsigned*)As;
            for (int i = threadIdx.x; i < 128 * 64; i += blockDim.x) {
                const int rr = i >> 6, c2 = (i & 63) << 1;
                const float a0 = agg[(size_t)(rowBase + rr) * 512 + kc * 128 + c2];
                const float a1 = agg[(size_t)(rowBase + rr) * 512 + kc * 128 + c2 + 1];
                const unsigned u0 = __builtin_bit_cast(unsigned short, (__bf16)a0);
                const unsigned u1 = __builtin_bit_cast(unsigned short, (__bf16)a1);
                As32[i] = u0 | (u1 << 16);
            }
        }
        // stage the Wt^T chunk: Bs[n][k'] = Wtt[n*512 + kc*128 + k']
        {
            const uint4* srcb = (const uint4*)Wtt;
            uint4*       dstb = (uint4*)Bs;
            for (int i = threadIdx.x; i < 2048; i += blockDim.x) {
                const int n = i >> 4, c = i & 15;
                dstb[i] = srcb[n * 64 + kc * 16 + c];
            }
        }
        __syncthreads();

        v16bf afrag[4];
#pragma unroll
        for (int kt = 0; kt < 4; ++kt)
            afrag[kt] = load_frag(As + m0 * 128 + kt * 32, 128);

#pragma unroll
        for (int nt = 0; nt < 8; ++nt) {
#pragma unroll
            for (int kt = 0; kt < 4; ++kt) {
                v16bf b = load_frag(Bs + (nt * 16) * 128 + kt * 32, 128);
                acc[nt] = __builtin_amdgcn_wmma_f32_16x16x32_bf16(false, afrag[kt], false, b,
                                                                  (short)0, acc[nt], false, false);
            }
        }
        __syncthreads();
    }

#pragma unroll
    for (int nt = 0; nt < 8; ++nt) {
        const float bv_ = bt[nt * 16 + nsub];
#pragma unroll
        for (int j = 0; j < 8; ++j)
            out[(size_t)(rowBase + m0 + mo + j) * 128 + nt * 16 + nsub] = acc[nt][j] + bv_;
    }
}

// ---------------- host launcher ----------------
static inline size_t align256(size_t x) { return (x + 255) & ~(size_t)255; }

extern "C" void kernel_launch(void* const* d_in, const int* in_sizes, int n_in,
                              void* d_out, int out_size, void* d_ws, size_t ws_size,
                              hipStream_t stream) {
    const float* h   = (const float*)d_in[0];
    const float* Wk  = (const float*)d_in[1];
    const float* bk  = (const float*)d_in[2];
    const float* Wq  = (const float*)d_in[3];
    const float* bq  = (const float*)d_in[4];
    const float* Wv  = (const float*)d_in[5];
    const float* bv  = (const float*)d_in[6];
    const float* Wt  = (const float*)d_in[7];
    const float* bt  = (const float*)d_in[8];
    const int*   src = (const int*)d_in[9];
    const int*   dst = (const int*)d_in[10];
    const int*   et  = (const int*)d_in[11];
    float* out = (float*)d_out;

    // workspace carve
    char* ws = (char*)d_ws;
    __bf16* h_bf  = (__bf16*)ws;  ws += align256((size_t)NN * 128 * 2);
    __bf16* W_bf  = (__bf16*)ws;  ws += align256((size_t)3 * RR * 128 * 128 * 2);
    __bf16* Wt_bf = (__bf16*)ws;  ws += align256((size_t)512 * 128 * 2);
    float*  KQV   = (float*)ws;   ws += align256((size_t)3 * RR * NN * 128 * 4);
    float*  score = (float*)ws;   ws += align256((size_t)EE * HH * 4);
    unsigned* smax = (unsigned*)ws; ws += align256((size_t)NN * RR * HH * 4);
    float*  den   = (float*)ws;   ws += align256((size_t)NN * RR * HH * 4);
    float*  agg   = (float*)ws;   ws += align256((size_t)NN * 512 * 4);

    const int B = 256;

    // 1) convert to bf16 (weights transposed to [N][K] for the B-side of WMMA)
    cvt_f32_bf16<<<2048, B, 0, stream>>>(h,  h_bf, NN * 128);
    cvt_w_t<<<320, B, 0, stream>>>(Wk, W_bf + 0 * RR * 128 * 128, RR);
    cvt_w_t<<<320, B, 0, stream>>>(Wq, W_bf + 1 * RR * 128 * 128, RR);
    cvt_w_t<<<320, B, 0, stream>>>(Wv, W_bf + 2 * RR * 128 * 128, RR);
    cvt_wt_t<<<256, B, 0, stream>>>(Wt, Wt_bf);

    // 2) zero accumulators (segmax ordered encoding: 0 == most-negative)
    hipMemsetAsync(smax, 0, (size_t)NN * RR * HH * 4, stream);
    hipMemsetAsync(den,  0, (size_t)NN * RR * HH * 4, stream);
    hipMemsetAsync(agg,  0, (size_t)NN * 512 * 4, stream);

    // 3) K/Q/V projections (WMMA)
    proj_gemm<<<3 * RR * (NN / 128), B, 0, stream>>>(h_bf, W_bf, bk, bq, bv, KQV);

    // 4) edge attention scores + segment max
    edge_scores<<<(EE * HH) / B, B, 0, stream>>>(KQV, src, dst, et, score, smax);

    // 5) exp + denominators
    edge_exp<<<(EE * HH) / B, B, 0, stream>>>(dst, et, score, smax, den);

    // 6) weighted aggregation
    edge_agg<<<(EE * 128) / B, B, 0, stream>>>(KQV, src, dst, et, score, den, agg);

    // 7) output projection (WMMA)
    out_gemm<<<NN / 128, B, 0, stream>>>(agg, Wt_bf, bt, out);
}
